// ChamferLoss_50190987821461
// MI455X (gfx1250) — compile-verified, matrix operand order checked by reading, not run on hardware
//
#include <hip/hip_runtime.h>

typedef __attribute__((ext_vector_type(16))) _Float16 v16h;
typedef __attribute__((ext_vector_type(8)))  _Float16 v8h;
typedef __attribute__((ext_vector_type(8)))  float    v8f;

#define B_  32
#define N_  2048
#define M_  2048
#define D_  32
#define ROWS_PER_BLOCK 128   // 8 waves x 16 rows

// ---------------------------------------------------------------------------
// Init: +inf keys for column minima, zero the scalar output accumulator.
// ---------------------------------------------------------------------------
__global__ void chamfer_init(unsigned* __restrict__ gcol, float* __restrict__ out) {
  int i = blockIdx.x * blockDim.x + threadIdx.x;
  if (i == 0) out[0] = 0.0f;
  const int total = B_ * M_;
  const int stride = gridDim.x * blockDim.x;
  for (; i < total; i += stride) gcol[i] = 0x7F800000u;  // bits of +inf
}

// ---------------------------------------------------------------------------
// Prep: one pass over X and Y. Convert f32 -> f16 (row-major, 64B rows) and
// compute squared norms in f32 (exact). Done ONCE, so the WMMA hot loop has
// zero conversion / norm VALU work.
// ---------------------------------------------------------------------------
__global__ __launch_bounds__(256) void chamfer_prep(
    const float* __restrict__ X, const float* __restrict__ Y,
    _Float16* __restrict__ Xh, _Float16* __restrict__ Yh,
    float* __restrict__ x2, float* __restrict__ y2)
{
  const int totalX = B_ * N_;
  const int total  = totalX + B_ * M_;
  const int stride = gridDim.x * blockDim.x;
  for (int i = blockIdx.x * blockDim.x + threadIdx.x; i < total; i += stride) {
    const float* src;
    _Float16*    dst;
    float*       nrm;
    if (i < totalX) {
      src = X + (size_t)i * D_;  dst = Xh + (size_t)i * D_;  nrm = x2 + i;
    } else {
      int j = i - totalX;
      src = Y + (size_t)j * D_;  dst = Yh + (size_t)j * D_;  nrm = y2 + j;
    }
    float s = 0.0f;
    alignas(16) _Float16 h[D_];
#pragma unroll
    for (int d = 0; d < D_; d += 4) {
      float4 v = *(const float4*)(src + d);
      s = fmaf(v.x, v.x, s); s = fmaf(v.y, v.y, s);
      s = fmaf(v.z, v.z, s); s = fmaf(v.w, v.w, s);
      h[d]   = (_Float16)v.x; h[d+1] = (_Float16)v.y;
      h[d+2] = (_Float16)v.z; h[d+3] = (_Float16)v.w;
    }
    const uint4* hv = (const uint4*)h;
    uint4* dv = (uint4*)dst;
#pragma unroll
    for (int q = 0; q < 4; ++q) dv[q] = hv[q];
    *nrm = s;
  }
}

// ---------------------------------------------------------------------------
// Main: per (batch, 128-row block). Each wave owns a 16x32 f16 A fragment and
// sweeps all 128 Y tiles. The norm bias is folded into the WMMA C operand:
//   s = X.Y^T - (x2+y2)/2   =>   d2 = -2*s,   min d2 == max s.
// So the post-WMMA epilogue is pure max-accumulation; the cinit pk_fma's sit
// BEFORE the WMMA under the load wait. Software-pipelined (WMMA i+1 issues
// before epilogue i) to fill the WMMA->VALU hazard slots; unroll 2 so the
// c/cinit rotation ping-pongs registers instead of copying.
// ---------------------------------------------------------------------------
__global__ __launch_bounds__(256) void chamfer_main(
    const _Float16* __restrict__ Xh, const _Float16* __restrict__ Yh,
    const float* __restrict__ x2, const float* __restrict__ y2,
    float* __restrict__ growd,      // [B_*N_] sqrt'd row minima (exclusive writes)
    unsigned* __restrict__ gcol)    // [B_*M_] clamped d2 bit-keys (atomicMin)
{
  __shared__ unsigned lcol[M_];     // 8 KB: per-block column minima keys

  const int tid    = threadIdx.x;
  const int lane   = tid & 31;      // wave32
  const int wave   = tid >> 5;
  const int lane15 = lane & 15;
  const int half   = lane >> 4;     // 0: lanes 0-15, 1: lanes 16-31
  const int b      = blockIdx.y;
  const int nbase  = blockIdx.x * ROWS_PER_BLOCK + wave * 16;

  for (int i = tid; i < M_; i += 256) lcol[i] = 0x7F800000u;
  __syncthreads();

  // ---- A fragment (16x32 f16): lane half selects K subsets {kb..kb+7, 16+kb..} ----
  const int kb = half ? 8 : 0;
  const _Float16* px = Xh + ((size_t)b * N_ + (size_t)(nbase + lane15)) * D_;
  v8h a0 = *(const v8h*)(px + kb);
  v8h a1 = *(const v8h*)(px + 16 + kb);
  v16h afrag;
#pragma unroll
  for (int i = 0; i < 8; ++i) { afrag[i] = a0[i]; afrag[8 + i] = a1[i]; }

  // ---- -x^2/2 for the 8 C-rows this lane sees (broadcast float4 loads) ----
  const float* px2 = x2 + (size_t)b * N_ + (size_t)nbase + 8 * half;
  float4 xa = *(const float4*)(px2);
  float4 xb = *(const float4*)(px2 + 4);
  float mx2[8] = {xa.x, xa.y, xa.z, xa.w, xb.x, xb.y, xb.z, xb.w};
#pragma unroll
  for (int r = 0; r < 8; ++r) mx2[r] *= -0.5f;

  float rowmax[8];
#pragma unroll
  for (int r = 0; r < 8; ++r) rowmax[r] = -3.0e38f;

  const _Float16* pyb = Yh + (size_t)b * M_ * D_ + (size_t)lane15 * D_;
  const float*    y2b = y2 + (size_t)b * M_;

  // Build the C operand for one tile: cinit[r] = -(x2[r] + y2[col])/2
  auto make_cinit = [&](float y2cc) {
    v8f ci;
#pragma unroll
    for (int r = 0; r < 8; ++r) ci[r] = fmaf(-0.5f, y2cc, mx2[r]);
    return ci;
  };
  // Epilogue for one finished tile: pure max-accumulation of s = -d2/2.
  auto epilogue = [&](v8f cc, int mbb) {
    float colmax = -3.0e38f;
#pragma unroll
    for (int r = 0; r < 8; ++r) {
      rowmax[r] = fmaxf(rowmax[r], cc[r]);
      colmax    = fmaxf(colmax, cc[r]);
    }
    float d2c = fmaxf(-2.0f * colmax, 0.0f);
    // both halves hit the same slot; ds_min_u32 does the cross-half combine
    atomicMin(&lcol[mbb + lane15], __float_as_uint(d2c));
  };

  // ---- pipeline prologue: tile 0 ----
  v8f c;
  {
    v8h b0 = *(const v8h*)(pyb + kb);
    v8h b1 = *(const v8h*)(pyb + 16 + kb);
    v16h bfrag;
#pragma unroll
    for (int i = 0; i < 8; ++i) { bfrag[i] = b0[i]; bfrag[8 + i] = b1[i]; }
    v8f ci = make_cinit(y2b[lane15]);
    c = __builtin_amdgcn_wmma_f32_16x16x32_f16(false, afrag, false, bfrag,
                                               (short)0, ci, false, false);
  }

  // ---- steady state: loads + cinit + WMMA(i), then epilogue(i-1) ----
#pragma unroll 2
  for (int mb = 16; mb < M_; mb += 16) {
    const _Float16* py = pyb + (size_t)mb * D_;
    v8h b0 = *(const v8h*)(py + kb);
    v8h b1 = *(const v8h*)(py + 16 + kb);
    float y2n = y2b[mb + lane15];
    __builtin_prefetch(py + 32 * D_, 0, 3);      // prefetch 2 tiles ahead

    v16h bfrag;
#pragma unroll
    for (int i = 0; i < 8; ++i) { bfrag[i] = b0[i]; bfrag[8 + i] = b1[i]; }
    v8f ci = make_cinit(y2n);                    // pk_fma's hide under load wait
    v8f cn = __builtin_amdgcn_wmma_f32_16x16x32_f16(false, afrag, false, bfrag,
                                                    (short)0, ci, false, false);
    epilogue(c, mb - 16);                        // independent VALU covers WMMA hazard
    c = cn;
  }
  epilogue(c, M_ - 16);                          // pipeline drain

  // ---- finish row minima: reduce max(s) across the 16 lanes of each half ----
#pragma unroll
  for (int r = 0; r < 8; ++r) {
    float v = rowmax[r];
    v = fmaxf(v, __shfl_xor(v, 1, 32));
    v = fmaxf(v, __shfl_xor(v, 2, 32));
    v = fmaxf(v, __shfl_xor(v, 4, 32));
    v = fmaxf(v, __shfl_xor(v, 8, 32));
    rowmax[r] = v;
  }
  if (lane15 == 0) {
#pragma unroll
    for (int r = 0; r < 8; ++r) {
      int row = nbase + r + 8 * half;
      float d2 = fmaxf(-2.0f * rowmax[r], 0.0f);
      growd[(size_t)b * N_ + row] = sqrtf(d2);
    }
  }

  __syncthreads();
  unsigned* gc = gcol + (size_t)b * M_;
  for (int i = tid; i < M_; i += 256) atomicMin(&gc[i], lcol[i]);
}

// ---------------------------------------------------------------------------
// Finalize: mean over sqrt'd minima. result = (mean X_inf + mean Y_inf)/2,
// averaged over batches (N==M so weights are uniform per array).
// ---------------------------------------------------------------------------
__global__ __launch_bounds__(256) void chamfer_finalize(
    const float* __restrict__ growd, const unsigned* __restrict__ gcol,
    float* __restrict__ out)
{
  __shared__ float red[256];
  const float wr = 1.0f / (2.0f * (float)B_ * (float)N_);
  const float wc = 1.0f / (2.0f * (float)B_ * (float)M_);
  const int stride = gridDim.x * blockDim.x;

  float acc = 0.0f;
  for (int i = blockIdx.x * blockDim.x + threadIdx.x; i < B_ * N_; i += stride)
    acc = fmaf(growd[i], wr, acc);
  for (int i = blockIdx.x * blockDim.x + threadIdx.x; i < B_ * M_; i += stride)
    acc = fmaf(sqrtf(__uint_as_float(gcol[i])), wc, acc);

  red[threadIdx.x] = acc;
  __syncthreads();
  for (int s = 128; s > 0; s >>= 1) {
    if (threadIdx.x < s) red[threadIdx.x] += red[threadIdx.x + s];
    __syncthreads();
  }
  if (threadIdx.x == 0) atomicAdd(out, red[0]);
}

// ---------------------------------------------------------------------------
extern "C" void kernel_launch(void* const* d_in, const int* in_sizes, int n_in,
                              void* d_out, int out_size, void* d_ws, size_t ws_size,
                              hipStream_t stream) {
  const float* X = (const float*)d_in[0];   // [B, N, D] f32
  const float* Y = (const float*)d_in[1];   // [B, M, D] f32
  float* out = (float*)d_out;               // scalar f32

  char* ws = (char*)d_ws;
  _Float16* Xh    = (_Float16*)ws;                       ws += (size_t)B_ * N_ * D_ * 2; // 4 MB
  _Float16* Yh    = (_Float16*)ws;                       ws += (size_t)B_ * M_ * D_ * 2; // 4 MB
  float*    x2    = (float*)ws;                          ws += (size_t)B_ * N_ * 4;      // 256 KB
  float*    y2    = (float*)ws;                          ws += (size_t)B_ * M_ * 4;      // 256 KB
  float*    growd = (float*)ws;                          ws += (size_t)B_ * N_ * 4;      // 256 KB
  unsigned* gcol  = (unsigned*)ws;                                                       // 256 KB

  chamfer_init<<<64, 256, 0, stream>>>(gcol, out);
  chamfer_prep<<<256, 256, 0, stream>>>(X, Y, Xh, Yh, x2, y2);

  dim3 grid(N_ / ROWS_PER_BLOCK, B_);       // (16, 32) blocks, 8 waves each
  chamfer_main<<<grid, 256, 0, stream>>>(Xh, Yh, x2, y2, growd, gcol);

  chamfer_finalize<<<128, 256, 0, stream>>>(growd, gcol, out);
}
